// GRU_Adversarial_22952305229977
// MI455X (gfx1250) — matrix operand order repaired and checked by reading, not compiled, not run on hardware
//
#include <hip/hip_runtime.h>
#include <math.h>

// ---------------------------------------------------------------------------
// Problem constants (from the reference)
// ---------------------------------------------------------------------------
#define T_ 128
#define B_ 256
#define H_ 512
#define G3_ 1536          // 3*H
#define IN_ 6
#define OUT_ 512          // N_STEPS * DIM_X
#define DX_ 4

typedef __bf16 v16bf __attribute__((ext_vector_type(16)));
typedef float  v8f   __attribute__((ext_vector_type(8)));

union AF { v16bf v; uint4 u[2]; };
union CF { v8f v; float f[8]; };

#define WMMA_BF16(A, Bf, C) \
    __builtin_amdgcn_wmma_f32_16x16x32_bf16(false, (A), false, (Bf), (short)0, (C), false, false)

__device__ __forceinline__ float sigm(float x) { return 1.0f / (1.0f + __expf(-x)); }

// ---------------------------------------------------------------------------
// Weight bf16 convert (no transpose): W[rows][cols] f32 -> Wt[rows][cols] bf16
// ---------------------------------------------------------------------------
__global__ void transpose_bf16_kernel(const float* __restrict__ W,
                                      __bf16* __restrict__ Wt,
                                      int rows, int cols) {
    int i = blockIdx.x * 256 + threadIdx.x;
    if (i >= rows * cols) return;
    int j = i / cols;
    int k = i - j * cols;
    Wt[i] = (__bf16)W[i];
}

// ---------------------------------------------------------------------------
// Initialize recurrent state from the 'hidden' input (f32 + bf16 mirrors)
// ---------------------------------------------------------------------------
__global__ void init_h_kernel(const float* __restrict__ hidden,
                              float* __restrict__ h0f, __bf16* __restrict__ h0b,
                              float* __restrict__ h1f, __bf16* __restrict__ h1b) {
    int i = blockIdx.x * 256 + threadIdx.x;   // 0 .. B*H-1
    float a = hidden[i];
    float c = hidden[(size_t)B_ * H_ + i];
    h0f[i] = a; h0b[i] = (__bf16)a;
    h1f[i] = c; h1b[i] = (__bf16)c;
}

// ---------------------------------------------------------------------------
// GRU layer 0, one time step.  grid = (H/64, B/32), block = 256 (8 waves).
// Wave (wm, wn) computes a 16x16 tile of each gate pre-activation via
// v_wmma_f32_16x16x32_bf16, then applies the GRU cell.
// gi (K=6 input matmul) is folded in as a scalar 6-MAC dot product.
// ---------------------------------------------------------------------------
__global__ void __launch_bounds__(256)
gru0_step_kernel(const __bf16* __restrict__ h_in_b,
                 const float*  __restrict__ h_in_f,
                 float*        __restrict__ h_out_f,
                 __bf16*       __restrict__ h_out_b,
                 __bf16*       __restrict__ y0,        // [T][B][H]
                 const __bf16* __restrict__ wt_hh,     // [3H][H] (not transposed)
                 const float*  __restrict__ pN,        // [T][B][4]
                 const float*  __restrict__ mN,        // [T][B][2]
                 const float*  __restrict__ W_ih0,     // [3H][6]
                 const float*  __restrict__ b_ih0,
                 const float*  __restrict__ b_hh0,
                 int t) {
    const int tid  = threadIdx.x;
    const int wave = tid >> 5, lane = tid & 31;
    const int wm = wave & 1, wn = wave >> 1;          // 2 x 4 wave grid
    const int m0 = blockIdx.y * 32 + wm * 16;         // batch tile
    const int n0 = blockIdx.x * 64 + wn * 16;         // hidden tile
    const int lhalf = (lane & 16) ? 8 : 0;            // A k sub-base / C m offset
    const int l15 = lane & 15;

    v8f accR = {}, accZ = {}, accN = {};
    const __bf16* aRow = h_in_b + (size_t)(m0 + l15) * H_;
    for (int k0 = 0; k0 < H_; k0 += 32) {
        AF a, b;
        a.u[0] = *(const uint4*)(aRow + k0 + lhalf);
        a.u[1] = *(const uint4*)(aRow + k0 + lhalf + 16);
        const __bf16* bRow = wt_hh + (size_t)(n0 + l15) * H_ + k0 + lhalf;
        __builtin_prefetch(bRow + 32, 0, 0);
        b.u[0] = *(const uint4*)(bRow);
        b.u[1] = *(const uint4*)(bRow + 16);
        accR = WMMA_BF16(a.v, b.v, accR);
        b.u[0] = *(const uint4*)(bRow + (size_t)H_ * H_);
        b.u[1] = *(const uint4*)(bRow + (size_t)H_ * H_ + 16);
        accZ = WMMA_BF16(a.v, b.v, accZ);
        b.u[0] = *(const uint4*)(bRow + (size_t)2 * H_ * H_);
        b.u[1] = *(const uint4*)(bRow + (size_t)2 * H_ * H_ + 16);
        accN = WMMA_BF16(a.v, b.v, accN);
    }

    const int ng = n0 + l15;                           // hidden unit 0..511
    const float bhr = b_hh0[ng], bhz = b_hh0[H_ + ng], bhn = b_hh0[2 * H_ + ng];
    const float bir = b_ih0[ng], biz = b_ih0[H_ + ng], bin_ = b_ih0[2 * H_ + ng];
    float wr[IN_], wz[IN_], wnn[IN_];
#pragma unroll
    for (int k = 0; k < IN_; ++k) {
        wr[k]  = W_ih0[(size_t)ng * IN_ + k];
        wz[k]  = W_ih0[(size_t)(H_ + ng) * IN_ + k];
        wnn[k] = W_ih0[(size_t)(2 * H_ + ng) * IN_ + k];
    }
    CF cr, cz, cn; cr.v = accR; cz.v = accZ; cn.v = accN;
#pragma unroll
    for (int i = 0; i < 8; ++i) {
        const int m = m0 + lhalf + i;                  // batch index
        const float* xp = pN + ((size_t)t * B_ + m) * 4;
        const float* xm = mN + ((size_t)t * B_ + m) * 2;
        float x0 = xp[0], x1 = xp[1], x2 = xp[2], x3 = xp[3], x4 = xm[0], x5 = xm[1];
        float gir = bir  + x0*wr[0] + x1*wr[1] + x2*wr[2] + x3*wr[3] + x4*wr[4] + x5*wr[5];
        float giz = biz  + x0*wz[0] + x1*wz[1] + x2*wz[2] + x3*wz[3] + x4*wz[4] + x5*wz[5];
        float gin = bin_ + x0*wnn[0]+ x1*wnn[1]+ x2*wnn[2]+ x3*wnn[3]+ x4*wnn[4]+ x5*wnn[5];
        float r  = sigm(gir + cr.f[i] + bhr);
        float z  = sigm(giz + cz.f[i] + bhz);
        float nn = tanhf(gin + r * (cn.f[i] + bhn));
        float ho = h_in_f[(size_t)m * H_ + ng];
        float hn = (1.0f - z) * nn + z * ho;
        h_out_f[(size_t)m * H_ + ng] = hn;
        __bf16 hb = (__bf16)hn;
        h_out_b[(size_t)m * H_ + ng] = hb;
        y0[((size_t)t * B_ + m) * H_ + ng] = hb;
    }
}

// ---------------------------------------------------------------------------
// GRU layer 1, one time step.  Same tiling; both gi (from y0[t]) and gh
// (from h) are K=512 WMMA matmuls sharing the K loop.
// ---------------------------------------------------------------------------
__global__ void __launch_bounds__(256)
gru1_step_kernel(const __bf16* __restrict__ h_in_b,
                 const float*  __restrict__ h_in_f,
                 float*        __restrict__ h_out_f,
                 __bf16*       __restrict__ h_out_b,
                 const __bf16* __restrict__ y0,        // [T][B][H]
                 __bf16*       __restrict__ y1,        // [T][B][H]
                 const __bf16* __restrict__ wt_hh,     // [3H][H]
                 const __bf16* __restrict__ wt_ih,     // [3H][H]
                 const float*  __restrict__ b_ih1,
                 const float*  __restrict__ b_hh1,
                 int t) {
    const int tid  = threadIdx.x;
    const int wave = tid >> 5, lane = tid & 31;
    const int wm = wave & 1, wn = wave >> 1;
    const int m0 = blockIdx.y * 32 + wm * 16;
    const int n0 = blockIdx.x * 64 + wn * 16;
    const int lhalf = (lane & 16) ? 8 : 0;
    const int l15 = lane & 15;

    v8f hR = {}, hZ = {}, hN = {}, iR = {}, iZ = {}, iN = {};
    const __bf16* aH = h_in_b + (size_t)(m0 + l15) * H_;
    const __bf16* aX = y0 + ((size_t)t * B_ + m0 + l15) * H_;
    for (int k0 = 0; k0 < H_; k0 += 32) {
        AF a, ax, b;
        a.u[0]  = *(const uint4*)(aH + k0 + lhalf);
        a.u[1]  = *(const uint4*)(aH + k0 + lhalf + 16);
        ax.u[0] = *(const uint4*)(aX + k0 + lhalf);
        ax.u[1] = *(const uint4*)(aX + k0 + lhalf + 16);
        const __bf16* bh = wt_hh + (size_t)(n0 + l15) * H_ + k0 + lhalf;
        const __bf16* bi = wt_ih + (size_t)(n0 + l15) * H_ + k0 + lhalf;
        __builtin_prefetch(bh + 32, 0, 0);
        __builtin_prefetch(bi + 32, 0, 0);
        b.u[0] = *(const uint4*)(bh);           b.u[1] = *(const uint4*)(bh + 16);
        hR = WMMA_BF16(a.v, b.v, hR);
        b.u[0] = *(const uint4*)(bh + (size_t)H_ * H_);      b.u[1] = *(const uint4*)(bh + (size_t)H_ * H_ + 16);
        hZ = WMMA_BF16(a.v, b.v, hZ);
        b.u[0] = *(const uint4*)(bh + (size_t)2 * H_ * H_);  b.u[1] = *(const uint4*)(bh + (size_t)2 * H_ * H_ + 16);
        hN = WMMA_BF16(a.v, b.v, hN);
        b.u[0] = *(const uint4*)(bi);           b.u[1] = *(const uint4*)(bi + 16);
        iR = WMMA_BF16(ax.v, b.v, iR);
        b.u[0] = *(const uint4*)(bi + (size_t)H_ * H_);      b.u[1] = *(const uint4*)(bi + (size_t)H_ * H_ + 16);
        iZ = WMMA_BF16(ax.v, b.v, iZ);
        b.u[0] = *(const uint4*)(bi + (size_t)2 * H_ * H_);  b.u[1] = *(const uint4*)(bi + (size_t)2 * H_ * H_ + 16);
        iN = WMMA_BF16(ax.v, b.v, iN);
    }

    const int ng = n0 + l15;
    const float bhr = b_hh1[ng], bhz = b_hh1[H_ + ng], bhn = b_hh1[2 * H_ + ng];
    const float bir = b_ih1[ng], biz = b_ih1[H_ + ng], bin_ = b_ih1[2 * H_ + ng];
    CF chr_, chz, chn, cir, ciz, cin_;
    chr_.v = hR; chz.v = hZ; chn.v = hN; cir.v = iR; ciz.v = iZ; cin_.v = iN;
#pragma unroll
    for (int i = 0; i < 8; ++i) {
        const int m = m0 + lhalf + i;
        float r  = sigm(cir.f[i] + bir + chr_.f[i] + bhr);
        float z  = sigm(ciz.f[i] + biz + chz.f[i] + bhz);
        float nn = tanhf(cin_.f[i] + bin_ + r * (chn.f[i] + bhn));
        float ho = h_in_f[(size_t)m * H_ + ng];
        float hn = (1.0f - z) * nn + z * ho;
        h_out_f[(size_t)m * H_ + ng] = hn;
        __bf16 hb = (__bf16)hn;
        h_out_b[(size_t)m * H_ + ng] = hb;
        y1[((size_t)t * B_ + m) * H_ + ng] = hb;
    }
}

// ---------------------------------------------------------------------------
// u = y1 @ W_lin.T + b_lin over all T*B rows.  grid = (OUT/64, T*B/64),
// block = 256 (8 waves, 4x2); each wave computes a 16x32 f32 tile.
// ---------------------------------------------------------------------------
__global__ void __launch_bounds__(256)
linear_u_kernel(const __bf16* __restrict__ y1,        // [T*B][H]
                const __bf16* __restrict__ wt_lin,    // [OUT][H]
                const float*  __restrict__ b_lin,
                float*        __restrict__ u) {       // [T*B][OUT]
    const int tid  = threadIdx.x;
    const int wave = tid >> 5, lane = tid & 31;
    const int wm = wave >> 1, wn = wave & 1;          // 4 x 2 wave grid
    const int m0 = blockIdx.y * 64 + wm * 16;
    const int n0 = blockIdx.x * 64 + wn * 32;
    const int lhalf = (lane & 16) ? 8 : 0;
    const int l15 = lane & 15;

    v8f c0 = {}, c1 = {};
    const __bf16* aRow = y1 + (size_t)(m0 + l15) * H_;
    for (int k0 = 0; k0 < H_; k0 += 32) {
        AF a, b;
        a.u[0] = *(const uint4*)(aRow + k0 + lhalf);
        a.u[1] = *(const uint4*)(aRow + k0 + lhalf + 16);
        const __bf16* bRow = wt_lin + (size_t)(n0 + l15) * H_ + k0 + lhalf;
        __builtin_prefetch(bRow + 32, 0, 0);
        b.u[0] = *(const uint4*)(bRow);       b.u[1] = *(const uint4*)(bRow + 16);
        c0 = WMMA_BF16(a.v, b.v, c0);
        b.u[0] = *(const uint4*)(bRow + (size_t)16 * H_);  b.u[1] = *(const uint4*)(bRow + (size_t)16 * H_ + 16);
        c1 = WMMA_BF16(a.v, b.v, c1);
    }
    CF f0, f1; f0.v = c0; f1.v = c1;
    const float bl0 = b_lin[n0 + l15], bl1 = b_lin[n0 + 16 + l15];
#pragma unroll
    for (int i = 0; i < 8; ++i) {
        const size_t m = (size_t)(m0 + lhalf + i);
        u[m * OUT_ + n0 + l15]      = f0.f[i] + bl0;
        u[m * OUT_ + n0 + 16 + l15] = f1.f[i] + bl1;
    }
}

// ---------------------------------------------------------------------------
// Per-(k,b) energies: one wave per (k,b).  diag_e = |u[k,b,4k:4k+4]|^2,
// suffix = sum_{c>=4k} u[k,b,c]^2, ratio = 1 - diag_e/suffix.
// ---------------------------------------------------------------------------
__global__ void energy_kernel(const float* __restrict__ u,
                              float* __restrict__ ratio,
                              float* __restrict__ suf,
                              float* __restrict__ diagu) {
    const int wg = blockIdx.x * 8 + (threadIdx.x >> 5);   // 0 .. T*B-1
    const int lane = threadIdx.x & 31;
    const int k = wg >> 8;          // / B_
    const int b = wg & 255;
    const float* row = u + ((size_t)k * B_ + b) * OUT_;
    const int c0 = 4 * k;
    float s = 0.0f;
    for (int c = c0 + lane; c < OUT_; c += 32) { float v = row[c]; s += v * v; }
#pragma unroll
    for (int off = 16; off; off >>= 1) s += __shfl_xor(s, off, 32);
    float du = (lane < 4) ? row[c0 + lane] : 0.0f;
    float de = du * du;
#pragma unroll
    for (int off = 16; off; off >>= 1) de += __shfl_xor(de, off, 32);
    const size_t idx = (size_t)k * B_ + b;
    if (lane < 4) diagu[idx * 4 + lane] = du;
    if (lane == 0) {
        suf[idx]   = s;
        ratio[idx] = 1.0f - de / s;
    }
}

// ---------------------------------------------------------------------------
// Sequential cumprod scan over k (one thread per batch element) + final write
// of u_N into d_out[0 .. T*B*DX).
// ---------------------------------------------------------------------------
__global__ void scan_out_kernel(const float* __restrict__ ratio,
                                const float* __restrict__ suf,
                                const float* __restrict__ diagu,
                                float* __restrict__ out) {
    const int b = threadIdx.x;      // 0..255
    float L = (float)T_;
    for (int k = 0; k < T_; ++k) {
        const size_t idx = (size_t)k * B_ + b;
        const float sc = sqrtf(L / suf[idx]);
#pragma unroll
        for (int d = 0; d < DX_; ++d) out[idx * 4 + d] = diagu[idx * 4 + d] * sc;
        L *= ratio[idx];
    }
}

// ---------------------------------------------------------------------------
// hidden_out = stack(h0T, h1T) -> d_out[T*B*DX ...]
// ---------------------------------------------------------------------------
__global__ void copy_hidden_kernel(const float* __restrict__ h0,
                                   const float* __restrict__ h1,
                                   float* __restrict__ out) {
    const int i = blockIdx.x * 256 + threadIdx.x;   // 0 .. B*H-1
    out[i] = h0[i];
    out[(size_t)B_ * H_ + i] = h1[i];
}

// ---------------------------------------------------------------------------
// Host launcher
// ---------------------------------------------------------------------------
extern "C" void kernel_launch(void* const* d_in, const int* in_sizes, int n_in,
                              void* d_out, int out_size, void* d_ws, size_t ws_size,
                              hipStream_t stream) {
    const float* pN     = (const float*)d_in[0];
    const float* mN     = (const float*)d_in[1];
    const float* hidden = (const float*)d_in[2];
    const float* W_ih0  = (const float*)d_in[3];
    const float* W_hh0  = (const float*)d_in[4];
    const float* b_ih0  = (const float*)d_in[5];
    const float* b_hh0  = (const float*)d_in[6];
    const float* W_ih1  = (const float*)d_in[7];
    const float* W_hh1  = (const float*)d_in[8];
    const float* b_ih1  = (const float*)d_in[9];
    const float* b_hh1  = (const float*)d_in[10];
    const float* W_lin  = (const float*)d_in[11];
    const float* b_lin  = (const float*)d_in[12];
    float* out = (float*)d_out;

    // Workspace carve-up (all offsets multiples of 256B by construction).
    char* ws = (char*)d_ws;
    size_t off = 0;
    auto take = [&](size_t bytes) -> char* {
        char* p = ws + off;
        off += (bytes + 255) & ~(size_t)255;
        return p;
    };
    const size_t WT3 = (size_t)G3_ * H_ * sizeof(__bf16);
    __bf16* wt_hh0 = (__bf16*)take(WT3);
    __bf16* wt_hh1 = (__bf16*)take(WT3);
    __bf16* wt_ih1 = (__bf16*)take(WT3);
    __bf16* wt_lin = (__bf16*)take((size_t)H_ * OUT_ * sizeof(__bf16));
    float*  h0f[2]; __bf16* h0b[2]; float* h1f[2]; __bf16* h1b[2];
    for (int p = 0; p < 2; ++p) {
        h0f[p] = (float*) take((size_t)B_ * H_ * sizeof(float));
        h0b[p] = (__bf16*)take((size_t)B_ * H_ * sizeof(__bf16));
        h1f[p] = (float*) take((size_t)B_ * H_ * sizeof(float));
        h1b[p] = (__bf16*)take((size_t)B_ * H_ * sizeof(__bf16));
    }
    __bf16* y0  = (__bf16*)take((size_t)T_ * B_ * H_ * sizeof(__bf16));
    __bf16* y1  = (__bf16*)take((size_t)T_ * B_ * H_ * sizeof(__bf16));
    float*  u   = (float*) take((size_t)T_ * B_ * OUT_ * sizeof(float));
    float*  rat = (float*) take((size_t)T_ * B_ * sizeof(float));
    float*  suf = (float*) take((size_t)T_ * B_ * sizeof(float));
    float*  dgu = (float*) take((size_t)T_ * B_ * DX_ * sizeof(float));
    (void)ws_size; (void)in_sizes; (void)n_in; (void)out_size;

    // 1) bf16-convert weights (layout unchanged), init h state
    transpose_bf16_kernel<<<(G3_ * H_ + 255) / 256, 256, 0, stream>>>(W_hh0, wt_hh0, G3_, H_);
    transpose_bf16_kernel<<<(G3_ * H_ + 255) / 256, 256, 0, stream>>>(W_hh1, wt_hh1, G3_, H_);
    transpose_bf16_kernel<<<(G3_ * H_ + 255) / 256, 256, 0, stream>>>(W_ih1, wt_ih1, G3_, H_);
    transpose_bf16_kernel<<<(OUT_ * H_ + 255) / 256, 256, 0, stream>>>(W_lin, wt_lin, OUT_, H_);
    init_h_kernel<<<(B_ * H_) / 256, 256, 0, stream>>>(hidden, h0f[0], h0b[0], h1f[0], h1b[0]);

    // 2) layer-0 recurrence (ping-pong h buffers; T even -> final state in buf 0)
    const dim3 gruGrid(H_ / 64, B_ / 32);
    for (int t = 0; t < T_; ++t) {
        const int pi = t & 1, po = pi ^ 1;
        gru0_step_kernel<<<gruGrid, 256, 0, stream>>>(
            h0b[pi], h0f[pi], h0f[po], h0b[po], y0,
            wt_hh0, pN, mN, W_ih0, b_ih0, b_hh0, t);
    }
    // 3) layer-1 recurrence
    for (int t = 0; t < T_; ++t) {
        const int pi = t & 1, po = pi ^ 1;
        gru1_step_kernel<<<gruGrid, 256, 0, stream>>>(
            h1b[pi], h1f[pi], h1f[po], h1b[po], y0, y1,
            wt_hh1, wt_ih1, b_ih1, b_hh1, t);
    }

    // 4) output linear
    linear_u_kernel<<<dim3(OUT_ / 64, (T_ * B_) / 64), 256, 0, stream>>>(y1, wt_lin, b_lin, u);

    // 5) control-energy normalization
    energy_kernel<<<(T_ * B_) / 8, 256, 0, stream>>>(u, rat, suf, dgu);
    scan_out_kernel<<<1, B_, 0, stream>>>(rat, suf, dgu, out);

    // 6) hidden_out
    copy_hidden_kernel<<<(B_ * H_) / 256, 256, 0, stream>>>(h0f[0], h1f[0],
                                                            out + (size_t)T_ * B_ * DX_);
}
